// _UnimolPEEELayer_38070590112227
// MI455X (gfx1250) — matrix operand order repaired and checked by reading, not compiled, not run on hardware
//
#include <hip/hip_runtime.h>
#include <hip/hip_bf16.h>
#include <math.h>
#include <stdint.h>

// ---------------- problem constants (from reference) ----------------
#define LSEQ 512
#define BB   4
#define DD   512
#define HH   16
#define DH   32
#define PP   32
#define EEN  16
#define DFF  2048
#define MR   (LSEQ * BB)      // 2048 rows of the [L,B,*] tensors
#define QKVW (3 * DD)         // 1536
#define LDT  34               // LDS tile row stride in bf16 (17 dwords, conflict-free)

typedef __bf16 bf16_t;
typedef __attribute__((ext_vector_type(16))) __bf16 v16bf;
typedef __attribute__((ext_vector_type(8)))  float  v8f;
typedef __attribute__((ext_vector_type(4)))  unsigned u32x4;
typedef __attribute__((ext_vector_type(8)))  int      i32x8;
typedef __attribute__((ext_vector_type(4)))  int      i32x4;

#ifndef __has_builtin
#define __has_builtin(x) 0
#endif
#if __has_builtin(__builtin_amdgcn_tensor_load_to_lds) && \
    __has_builtin(__builtin_amdgcn_s_wait_tensorcnt)
#define USE_TDM 1
#else
#define USE_TDM 0
#endif
#if defined(__has_include)
#if __has_include(<hip/amd_detail/amd_gfx1250_TDM.h>)
#define TDM_ARGS6 1
#else
#define TDM_ARGS6 0
#endif
#else
#define TDM_ARGS6 0
#endif

// ---------------- bf16 <-> f32 helpers (RNE) ----------------
__device__ __forceinline__ bf16_t f2bf(float f) {
  union { float f; unsigned u; } v; v.f = f;
  unsigned r = (v.u + 0x7FFFu + ((v.u >> 16) & 1u)) >> 16;
  union { unsigned short s; bf16_t b; } o; o.s = (unsigned short)r;
  return o.b;
}
__device__ __forceinline__ float bf2f(bf16_t b) {
  union { unsigned short s; bf16_t b; } i; i.b = b;
  union { unsigned u; float f; } o; o.u = ((unsigned)i.s) << 16;
  return o.f;
}
__device__ __forceinline__ float gelu_f(float x) {
  return 0.5f * x * (1.0f + erff(x * 0.70710678118654752440f));
}

// ======================================================================
// Tile staging: every LDS tile is rows x 32 bf16, row stride LDT=34.
// TDM path: one wave issues tensor_load_to_lds (D# per ISA 8.3/8.4),
//           pad_enable inserts 1 dword per 16 dwords -> row stride 34 bf16.
// Fallback: all 256 threads stage with b64 global loads / b32 LDS stores.
// ======================================================================
#if USE_TDM
__device__ __forceinline__ void tdm_stage32(unsigned lds_off, const void* gsrc,
                                            long row_stride_elems, int tile_rows,
                                            int valid_rows) {
  const unsigned long long ga = (unsigned long long)(uintptr_t)gsrc;
  u32x4 g0;
  g0[0] = 1u;                                    // count=1 (user descriptor)
  g0[1] = lds_off;                               // lds_addr (bytes)
  g0[2] = (unsigned)(ga & 0xFFFFFFFFu);          // global_addr[31:0]
  g0[3] = (unsigned)((ga >> 32) & 0x01FFFFFFu) | (2u << 30);  // addr[56:32]|type=2
  i32x8 g1;
  // data_size=1 (2B) | pad_enable | pad_interval=3 (16 dw) | pad_amount=0 (1 dw)
  g1[0] = (int)((1u << 16) | (1u << 20) | (3u << 22));
  g1[1] = (int)(32u << 16);                      // tensor_dim0[15:0] = 32
  g1[2] = (int)(((unsigned)valid_rows & 0xFFFFu) << 16);  // tensor_dim1[15:0]
  g1[3] = (int)(32u << 16);                      // tile_dim0 = 32
  g1[4] = (int)((unsigned)tile_rows & 0xFFFFu);  // tile_dim1 ; tile_dim2=0
  g1[5] = (int)(unsigned)((unsigned long long)row_stride_elems & 0xFFFFFFFFu);
  g1[6] = (int)(unsigned)(((unsigned long long)row_stride_elems >> 32) & 0xFFFFu);
  g1[7] = 0;
  const i32x4 z4 = {0, 0, 0, 0};
#if TDM_ARGS6
  const i32x8 z8 = {0, 0, 0, 0, 0, 0, 0, 0};
  __builtin_amdgcn_tensor_load_to_lds(g0, g1, z4, z4, z8, 0);
#else
  __builtin_amdgcn_tensor_load_to_lds(g0, g1, z4, z4, 0);
#endif
}
#endif

__device__ __forceinline__ void stage32_fallback(bf16_t* dst, const bf16_t* src,
                                                 long row_stride_elems,
                                                 int tile_rows, int valid_rows) {
  for (int c = threadIdx.x; c < tile_rows * 4; c += 256) {
    const int r = c >> 2, q = c & 3;
    unsigned long long d0 = 0ull, d1 = 0ull;
    if (r < valid_rows) {
      const unsigned long long* s =
          (const unsigned long long*)(src + (size_t)r * row_stride_elems + q * 8);
      d0 = s[0];
      d1 = s[1];
    }
    unsigned* w = (unsigned*)(dst + r * LDT + q * 8);
    w[0] = (unsigned)d0; w[1] = (unsigned)(d0 >> 32);
    w[2] = (unsigned)d1; w[3] = (unsigned)(d1 >> 32);
  }
}

// stage two tiles (A-style and B-style) then barrier.  wave0/wave1 own the
// TDM issues; consumers are ordered by s_wait_tensorcnt + s_barrier.
__device__ __forceinline__ void stage_pair(bf16_t* dstA, const bf16_t* srcA,
                                           long strideA, int rowsA, int validA,
                                           bf16_t* dstB, const bf16_t* srcB,
                                           long strideB, int rowsB, int validB) {
#if USE_TDM
  const int wave = threadIdx.x >> 5;
  if (wave == 0) {
    tdm_stage32((unsigned)(uintptr_t)dstA, srcA, strideA, rowsA, validA);
    __builtin_amdgcn_s_wait_tensorcnt(0);
  } else if (wave == 1) {
    tdm_stage32((unsigned)(uintptr_t)dstB, srcB, strideB, rowsB, validB);
    __builtin_amdgcn_s_wait_tensorcnt(0);
  }
#else
  stage32_fallback(dstA, srcA, strideA, rowsA, validA);
  stage32_fallback(dstB, srcB, strideB, rowsB, validB);
#endif
  __syncthreads();
}

// ---------------- WMMA fragment loaders (ISA 7.12.2, wave32) -------------
// A: 16x32 bf16 from row-major LDS tile [m][k], 8 dword loads.
__device__ __forceinline__ v16bf load_a_frag(const bf16_t* As, int row0) {
  const int lane = threadIdx.x & 31;
  const int m  = row0 + (lane & 15);
  const int kb = (lane >> 4) << 4;  // 0 or 16
  union { v16bf v; unsigned u[8]; } f;
#pragma unroll
  for (int v = 0; v < 8; ++v) {
    const int k0 = kb + 2 * v;
    f.u[v] = *(const unsigned*)(As + m * LDT + k0);
  }
  return f.v;
}
// B: 32x16 bf16 from N-major LDS tile [n][k], 8 dword loads.
// lane<16: N=lane, K=0..15 ; lane>=16: N=lane-16, K=16..31
__device__ __forceinline__ v16bf load_b_frag_nm(const bf16_t* Bt, int col0) {
  const int lane = threadIdx.x & 31;
  const int n  = col0 + (lane & 15);
  const int kb = (lane >> 4) << 4;  // 0 or 16
  union { v16bf v; unsigned u[8]; } f;
#pragma unroll
  for (int e = 0; e < 8; ++e)
    f.u[e] = *(const unsigned*)(Bt + n * LDT + kb + 2 * e);
  return f.v;
}
// B from K-major LDS tile [k][n] (strided) -- only used for the tiny V tile.
__device__ __forceinline__ v16bf load_b_frag_km(const bf16_t* Bs, int col0) {
  const int lane = threadIdx.x & 31;
  const int n  = col0 + (lane & 15);
  const int kb = (lane >> 4) << 4;
  v16bf b;
#pragma unroll
  for (int e = 0; e < 16; ++e) b[e] = Bs[(kb + e) * LDT + n];
  return b;
}

__device__ __forceinline__ v8f wmma_bf16(v16bf a, v16bf b, v8f c) {
  return __builtin_amdgcn_wmma_f32_16x16x32_bf16(false, a, false, b, (short)0, c,
                                                 false, false);
}

// ======================================================================
// GEMM: C[M,N] = act(A[M,K] @ W^T[N,K]^T + bias + resid)
// 64x64 macro tile, 8 waves -> two 16x16 WMMA accumulators each.
// ======================================================================
template <int ACT, int OBF, int RESID>
__global__ __launch_bounds__(256) void gemm_bf16_kernel(
    const bf16_t* __restrict__ A, const bf16_t* __restrict__ Wt,
    const float* __restrict__ bias, const float* __restrict__ resid,
    float* __restrict__ Cf, bf16_t* __restrict__ Cb, int M, int N, int K) {
  constexpr int BM = 64, BN = 64, BK = 32;
  __shared__ bf16_t As[BM * LDT];
  __shared__ bf16_t Bt[BN * LDT];

  const int tid = threadIdx.x;
  const int wave = tid >> 5;
  const int rowBase = blockIdx.y * BM;
  const int colBase = blockIdx.x * BN;
  const int wr = wave >> 1;  // 0..3
  const int wc = wave & 1;   // 0..1
  const int validA = (M - rowBase < BM) ? (M - rowBase) : BM;

  v8f acc0 = {0.f, 0.f, 0.f, 0.f, 0.f, 0.f, 0.f, 0.f};
  v8f acc1 = acc0;

  for (int k0 = 0; k0 < K; k0 += BK) {
    stage_pair(As, A + (size_t)rowBase * K + k0, K, BM, validA,
               Bt, Wt + (size_t)colBase * K + k0, K, BN, BN);
    const v16bf af  = load_a_frag(As, wr * 16);
    const v16bf bf0 = load_b_frag_nm(Bt, wc * 32);
    const v16bf bf1 = load_b_frag_nm(Bt, wc * 32 + 16);
    acc0 = wmma_bf16(af, bf0, acc0);
    acc1 = wmma_bf16(af, bf1, acc1);
    __syncthreads();
  }

  const int lane = tid & 31;
  const int n_in = lane & 15;
  const int m_in = (lane >> 4) << 3;
#pragma unroll
  for (int half = 0; half < 2; ++half) {
    const v8f& acc = half ? acc1 : acc0;
    const int gn = colBase + wc * 32 + half * 16 + n_in;
#pragma unroll
    for (int v = 0; v < 8; ++v) {
      const int gm = rowBase + wr * 16 + m_in + v;
      if (gm >= M) continue;
      float val = acc[v];
      if (bias) val += bias[gn];
      if (RESID) val += resid[(size_t)gm * N + gn];
      if (ACT == 1) val = gelu_f(val);
      if (OBF) Cb[(size_t)gm * N + gn] = f2bf(val);
      else     Cf[(size_t)gm * N + gn] = val;
    }
  }
}

// ======================================================================
// t_raw[bh,i,j] = sum_d q[bh,i,d]*k[bh,j,d]  (K=Dh=32 -> one WMMA per tile)
// K-tile is staged in its natural [j][d] order == N-major for WMMA B.
// ======================================================================
__global__ __launch_bounds__(256) void qk_scores_kernel(
    const bf16_t* __restrict__ qkv, float* __restrict__ t_raw) {
  __shared__ bf16_t Qs[64 * LDT];
  __shared__ bf16_t Kt[64 * LDT];
  const int tid = threadIdx.x;
  const int bh = blockIdx.z;
  const int b = bh / HH, h = bh % HH;
  const int i0 = blockIdx.y * 64;
  const int j0 = blockIdx.x * 64;
  const long rs = (long)BB * QKVW;  // row stride in qkv

  stage_pair(Qs, qkv + ((size_t)i0 * BB + b) * QKVW + h * DH, rs, 64, 64,
             Kt, qkv + ((size_t)j0 * BB + b) * QKVW + DD + h * DH, rs, 64, 64);

  const int wave = tid >> 5;
  const int wr = wave >> 1, wc = wave & 1;
  const v16bf af  = load_a_frag(Qs, wr * 16);
  const v16bf bf0 = load_b_frag_nm(Kt, wc * 32);
  const v16bf bf1 = load_b_frag_nm(Kt, wc * 32 + 16);
  v8f z = {0.f, 0.f, 0.f, 0.f, 0.f, 0.f, 0.f, 0.f};
  const v8f acc0 = wmma_bf16(af, bf0, z);
  const v8f acc1 = wmma_bf16(af, bf1, z);

  const int lane = tid & 31;
  const int n_in = lane & 15;
  const int m_in = (lane >> 4) << 3;
  const size_t base = (size_t)bh * LSEQ * LSEQ;
#pragma unroll
  for (int v = 0; v < 8; ++v) {
    const int gi = i0 + wr * 16 + m_in + v;
    t_raw[base + (size_t)gi * LSEQ + (j0 + wc * 32 + n_in)]      = acc0[v];
    t_raw[base + (size_t)gi * LSEQ + (j0 + wc * 32 + 16 + n_in)] = acc1[v];
  }
}

// ======================================================================
// ao[bh,i,:] = attn[bh,i,:] @ v[bh,:,:]   (N = Dh = 32)
// ======================================================================
__global__ __launch_bounds__(256) void attn_v_kernel(
    const bf16_t* __restrict__ attn, const bf16_t* __restrict__ qkv,
    bf16_t* __restrict__ ao) {
  __shared__ bf16_t As[64 * LDT];
  __shared__ bf16_t Vs[32 * LDT];  // [j][d]  (K-major for WMMA B)
  const int tid = threadIdx.x;
  const int bh = blockIdx.z;
  const int b = bh / HH, h = bh % HH;
  const int i0 = blockIdx.y * 64;
  const int wave = tid >> 5;
  const int wr = wave >> 1;  // 8 waves cover 64x32 as 4x2 tiles of 16x16
  const int wc = wave & 1;

  v8f acc = {0.f, 0.f, 0.f, 0.f, 0.f, 0.f, 0.f, 0.f};
  const size_t abase = (size_t)bh * LSEQ * LSEQ;
  const long rs = (long)BB * QKVW;

  for (int k0 = 0; k0 < LSEQ; k0 += 32) {
    stage_pair(As, attn + abase + (size_t)i0 * LSEQ + k0, LSEQ, 64, 64,
               Vs, qkv + ((size_t)k0 * BB + b) * QKVW + 2 * DD + h * DH, rs, 32, 32);
    const v16bf af = load_a_frag(As, wr * 16);
    const v16bf bf = load_b_frag_km(Vs, wc * 16);
    acc = wmma_bf16(af, bf, acc);
    __syncthreads();
  }

  const int lane = tid & 31;
  const int n = wc * 16 + (lane & 15);
  const int mb = wr * 16 + ((lane >> 4) << 3);
#pragma unroll
  for (int v = 0; v < 8; ++v)
    ao[((size_t)(i0 + mb + v) * BB + b) * DD + h * DH + n] = f2bf(acc[v]);
}

// ======================================================================
// small pair-bias tables: out[p,b,h,l] = <emb_p chunk, qkv chunk> (len 32)
// ======================================================================
__global__ void pair_table_kernel(const bf16_t* __restrict__ embp, int NP,
                                  int eoff, const bf16_t* __restrict__ qkv,
                                  int qoff, float* __restrict__ out) {
  const int idx = blockIdx.x * blockDim.x + threadIdx.x;
  const int total = NP * BB * HH * LSEQ;
  if (idx >= total) return;
  const int l = idx % LSEQ;
  const int h = (idx / LSEQ) % HH;
  const int b = (idx / (LSEQ * HH)) % BB;
  const int p = idx / (LSEQ * HH * BB);
  const bf16_t* e = embp + (size_t)p * QKVW + eoff + h * DH;
  const bf16_t* k = qkv + ((size_t)l * BB + b) * QKVW + qoff + h * DH;
  float s = 0.f;
#pragma unroll
  for (int d = 0; d < DH; ++d) s += bf2f(e[d]) * bf2f(k[d]);
  out[idx] = s;
}

// ======================================================================
// LayerNorm (fp32 in) -> bf16 out. one block (256 thr) per row of 512.
// ======================================================================
__global__ __launch_bounds__(256) void ln_bf16_kernel(
    const float* __restrict__ X, const float* __restrict__ g,
    const float* __restrict__ be, bf16_t* __restrict__ Y) {
  __shared__ float red[256];
  const int row = blockIdx.x;
  const int tid = threadIdx.x;
  const float* x = X + (size_t)row * DD;
  const float v0 = x[tid], v1 = x[tid + 256];
  red[tid] = v0 + v1;
  __syncthreads();
  for (int st = 128; st > 0; st >>= 1) {
    if (tid < st) red[tid] += red[tid + st];
    __syncthreads();
  }
  const float mean = red[0] * (1.0f / DD);
  __syncthreads();
  const float d0 = v0 - mean, d1 = v1 - mean;
  red[tid] = d0 * d0 + d1 * d1;
  __syncthreads();
  for (int st = 128; st > 0; st >>= 1) {
    if (tid < st) red[tid] += red[tid + st];
    __syncthreads();
  }
  const float rstd = rsqrtf(red[0] * (1.0f / DD) + 1e-5f);
  Y[(size_t)row * DD + tid]       = f2bf(d0 * rstd * g[tid] + be[tid]);
  Y[(size_t)row * DD + tid + 256] = f2bf(d1 * rstd * g[tid + 256] + be[tid + 256]);
}

// ======================================================================
// fused: logits = (t_raw + 4 pair-gathers) * rsqrt(Dh) + edge
//        edge_out = logits ; attn = softmax(logits) in bf16
// ======================================================================
__global__ __launch_bounds__(256) void softmax_gather_kernel(
    const float* __restrict__ t_raw, const float* __restrict__ edge,
    const int* __restrict__ bdist, const int* __restrict__ etok,
    const float* __restrict__ A_pe, const float* __restrict__ A_ee,
    const float* __restrict__ B_pe, const float* __restrict__ B_ee,
    bf16_t* __restrict__ attn, float* __restrict__ edge_out) {
  const int i = blockIdx.x;
  const int bh = blockIdx.y;
  const int b = bh / HH, h = bh % HH;
  const int tid = threadIdx.x;
  __shared__ float sBpe[PP];
  __shared__ float sBee[EEN];
  __shared__ float red[256];
  if (tid < PP)  sBpe[tid] = B_pe[(((size_t)tid * BB + b) * HH + h) * LSEQ + i];
  if (tid < EEN) sBee[tid] = B_ee[(((size_t)tid * BB + b) * HH + h) * LSEQ + i];
  __syncthreads();

  const size_t rbase = ((size_t)bh * LSEQ + i) * LSEQ;
  float logit[2];
#pragma unroll
  for (int q = 0; q < 2; ++q) {
    const int j = tid + q * 256;
    const int bd = bdist[((size_t)i * LSEQ + j) * BB + b];
    const int et = etok[((size_t)i * LSEQ + j) * BB + b];
    float v = t_raw[rbase + j] +
              A_pe[(((size_t)bd * BB + b) * HH + h) * LSEQ + j] +
              A_ee[(((size_t)et * BB + b) * HH + h) * LSEQ + j] + sBpe[bd] +
              sBee[et];
    v *= 0.17677669529663687f;  // 1/sqrt(32)
    const float lg = v + edge[rbase + j];
    edge_out[rbase + j] = lg;
    logit[q] = lg;
  }
  red[tid] = fmaxf(logit[0], logit[1]);
  __syncthreads();
  for (int st = 128; st > 0; st >>= 1) {
    if (tid < st) red[tid] = fmaxf(red[tid], red[tid + st]);
    __syncthreads();
  }
  const float rowmax = red[0];
  __syncthreads();
  const float e0 = __expf(logit[0] - rowmax);
  const float e1 = __expf(logit[1] - rowmax);
  red[tid] = e0 + e1;
  __syncthreads();
  for (int st = 128; st > 0; st >>= 1) {
    if (tid < st) red[tid] += red[tid + st];
    __syncthreads();
  }
  const float inv = 1.0f / red[0];
  attn[rbase + tid]       = f2bf(e0 * inv);
  attn[rbase + tid + 256] = f2bf(e1 * inv);
}

__global__ void f32_to_bf16_kernel(const float* __restrict__ in,
                                   bf16_t* __restrict__ out, int n) {
  const int idx = blockIdx.x * blockDim.x + threadIdx.x;
  if (idx < n) out[idx] = f2bf(in[idx]);
}

// weights: in [K,N] f32 -> out [N,K] bf16 (transposed for N-major staging)
__global__ void cvt_transpose_kernel(const float* __restrict__ in,
                                     bf16_t* __restrict__ out, int K, int N) {
  const int idx = blockIdx.x * blockDim.x + threadIdx.x;
  if (idx >= K * N) return;
  const int k = idx / N, n = idx % N;
  out[(size_t)n * K + k] = f2bf(in[idx]);
}

// ======================================================================
extern "C" void kernel_launch(void* const* d_in, const int* in_sizes, int n_in,
                              void* d_out, int out_size, void* d_ws,
                              size_t ws_size, hipStream_t stream) {
  (void)in_sizes; (void)n_in; (void)out_size; (void)ws_size;
  const float* x      = (const float*)d_in[0];
  const float* edge   = (const float*)d_in[1];
  const int*   bdist  = (const int*)d_in[2];
  const int*   etok   = (const int*)d_in[3];
  const float* W_in   = (const float*)d_in[4];
  const float* b_in   = (const float*)d_in[5];
  const float* pe_emb = (const float*)d_in[6];
  const float* ee_emb = (const float*)d_in[7];
  const float* W_out  = (const float*)d_in[8];
  const float* b_out  = (const float*)d_in[9];
  const float* W1     = (const float*)d_in[10];
  const float* bb1    = (const float*)d_in[11];
  const float* W2     = (const float*)d_in[12];
  const float* bb2    = (const float*)d_in[13];
  const float* g1     = (const float*)d_in[14];
  const float* be1    = (const float*)d_in[15];
  const float* g2     = (const float*)d_in[16];
  const float* be2    = (const float*)d_in[17];

  float* x2_out   = (float*)d_out;                           // [2048,512]
  float* edge_out = (float*)d_out + (size_t)LSEQ * BB * DD;  // [64,512,512]

  char* ws = (char*)d_ws;
  size_t off = 0;
  auto take = [&](size_t bytes) -> char* {
    char* p = ws + off;
    off = (off + bytes + 255) & ~(size_t)255;
    return p;
  };
  bf16_t* WinT_bf  = (bf16_t*)take((size_t)QKVW * DD * 2);   // [1536,512]
  bf16_t* WoutT_bf = (bf16_t*)take((size_t)DD * DD * 2);     // [512,512]
  bf16_t* W1T_bf   = (bf16_t*)take((size_t)DFF * DD * 2);    // [2048,512]
  bf16_t* W2T_bf   = (bf16_t*)take((size_t)DD * DFF * 2);    // [512,2048]
  bf16_t* pe_bf    = (bf16_t*)take((size_t)PP * DD * 2);
  bf16_t* ee_bf    = (bf16_t*)take((size_t)EEN * DD * 2);
  bf16_t* h_bf     = (bf16_t*)take((size_t)MR * DD * 2);
  bf16_t* qkv_bf   = (bf16_t*)take((size_t)MR * QKVW * 2);
  bf16_t* pep_bf   = (bf16_t*)take((size_t)PP * QKVW * 2);
  bf16_t* eep_bf   = (bf16_t*)take((size_t)EEN * QKVW * 2);
  float*  t_raw    = (float*)take((size_t)BB * HH * LSEQ * LSEQ * 4);
  float*  A_pe     = (float*)take((size_t)PP * BB * HH * LSEQ * 4);
  float*  A_ee     = (float*)take((size_t)EEN * BB * HH * LSEQ * 4);
  float*  B_pe     = (float*)take((size_t)PP * BB * HH * LSEQ * 4);
  float*  B_ee     = (float*)take((size_t)EEN * BB * HH * LSEQ * 4);
  bf16_t* attn_bf  = (bf16_t*)take((size_t)BB * HH * LSEQ * LSEQ * 2);
  bf16_t* ao_bf    = (bf16_t*)take((size_t)MR * DD * 2);
  float*  x1       = (float*)take((size_t)MR * DD * 4);
  bf16_t* h2_bf    = (bf16_t*)take((size_t)MR * DD * 2);
  bf16_t* ffn1_bf  = (bf16_t*)take((size_t)MR * DFF * 2);

  auto cvtT = [&](const float* src, bf16_t* dst, int K, int N) {
    cvt_transpose_kernel<<<(K * N + 255) / 256, 256, 0, stream>>>(src, dst, K, N);
  };
  cvtT(W_in, WinT_bf, DD, QKVW);
  cvtT(W_out, WoutT_bf, DD, DD);
  cvtT(W1, W1T_bf, DD, DFF);
  cvtT(W2, W2T_bf, DFF, DD);
  f32_to_bf16_kernel<<<(PP * DD + 255) / 256, 256, 0, stream>>>(pe_emb, pe_bf, PP * DD);
  f32_to_bf16_kernel<<<(EEN * DD + 255) / 256, 256, 0, stream>>>(ee_emb, ee_bf, EEN * DD);

  ln_bf16_kernel<<<MR, 256, 0, stream>>>(x, g1, be1, h_bf);

  const dim3 blk(256);
  // qkv = ln(x) @ W_in + b_in  -> bf16
  gemm_bf16_kernel<0, 1, 0><<<dim3(QKVW / 64, MR / 64), blk, 0, stream>>>(
      h_bf, WinT_bf, b_in, nullptr, nullptr, qkv_bf, MR, QKVW, DD);
  // tiny projected embedding tables
  gemm_bf16_kernel<0, 1, 0><<<dim3(QKVW / 64, 1), blk, 0, stream>>>(
      pe_bf, WinT_bf, b_in, nullptr, nullptr, pep_bf, PP, QKVW, DD);
  gemm_bf16_kernel<0, 1, 0><<<dim3(QKVW / 64, 1), blk, 0, stream>>>(
      ee_bf, WinT_bf, b_in, nullptr, nullptr, eep_bf, EEN, QKVW, DD);

  // raw attention scores q.k^T
  qk_scores_kernel<<<dim3(LSEQ / 64, LSEQ / 64, BB * HH), blk, 0, stream>>>(
      qkv_bf, t_raw);

  // bucket tables: A_* = emb q-chunk vs k ; B_* = emb k-chunk vs q
  {
    int tot = PP * BB * HH * LSEQ;
    pair_table_kernel<<<(tot + 255) / 256, 256, 0, stream>>>(pep_bf, PP, 0,
                                                             qkv_bf, DD, A_pe);
    pair_table_kernel<<<(tot + 255) / 256, 256, 0, stream>>>(pep_bf, PP, DD,
                                                             qkv_bf, 0, B_pe);
    tot = EEN * BB * HH * LSEQ;
    pair_table_kernel<<<(tot + 255) / 256, 256, 0, stream>>>(eep_bf, EEN, 0,
                                                             qkv_bf, DD, A_ee);
    pair_table_kernel<<<(tot + 255) / 256, 256, 0, stream>>>(eep_bf, EEN, DD,
                                                             qkv_bf, 0, B_ee);
  }

  // fused gather + scale + edge + softmax ; writes edge_out and attn(bf16)
  softmax_gather_kernel<<<dim3(LSEQ, BB * HH), blk, 0, stream>>>(
      t_raw, edge, bdist, etok, A_pe, A_ee, B_pe, B_ee, attn_bf, edge_out);

  // ao = attn @ v  -> bf16 in [L,B,D] layout
  attn_v_kernel<<<dim3(1, LSEQ / 64, BB * HH), blk, 0, stream>>>(attn_bf,
                                                                 qkv_bf, ao_bf);

  // x1 = x + ao @ W_out + b_out
  gemm_bf16_kernel<0, 0, 1><<<dim3(DD / 64, MR / 64), blk, 0, stream>>>(
      ao_bf, WoutT_bf, b_out, x, x1, nullptr, MR, DD, DD);

  ln_bf16_kernel<<<MR, 256, 0, stream>>>(x1, g2, be2, h2_bf);

  // ffn1 = gelu(h2 @ W1 + bb1) -> bf16
  gemm_bf16_kernel<1, 1, 0><<<dim3(DFF / 64, MR / 64), blk, 0, stream>>>(
      h2_bf, W1T_bf, bb1, nullptr, nullptr, ffn1_bf, MR, DFF, DD);

  // x2 = x1 + ffn1 @ W2 + bb2  -> first segment of d_out
  gemm_bf16_kernel<0, 0, 1><<<dim3(DD / 64, MR / 64), blk, 0, stream>>>(
      ffn1_bf, W2T_bf, bb2, x1, x2_out, nullptr, MR, DD, DFF);
}
